// SpatialShiftAttention_9723805958798
// MI455X (gfx1250) — matrix operand order repaired and checked by reading, not compiled, not run on hardware
//
#include <hip/hip_runtime.h>
#include <hip/hip_bf16.h>

// ---------------------------------------------------------------------------
// SpatialShiftAttention for MI455X (gfx1250, wave32, WMMA).
// y (50176 x 1152) staged in bf16 in d_ws; both big GEMMs use
// v_wmma_f32_16x16x32_bf16 with f32 accumulation, LDS tiles stored in
// per-lane fragment order (b128 LDS loads, no packing movs), double-buffered.
// ~0.5 GB total traffic -> ~21us at 23.3 TB/s; bf16 WMMA keeps math off the
// critical path.
// ---------------------------------------------------------------------------

typedef __attribute__((ext_vector_type(16))) __bf16 v16bf;
typedef __attribute__((ext_vector_type(8)))  __bf16 v8bf;
typedef __attribute__((ext_vector_type(8)))  float  v8f;

#define NTOK   50176   // 16*56*56
#define CIN    384
#define NFK    1152    // NF*K
#define NF     384
#define HH     56
#define WW     56
#define HW     3136
#define TILE_M 128
#define TILE_N 128
#define TILE_K 32
#define KT     12      // 384 / 32 k-tiles (both GEMMs)
#define LDA    40      // row stride (bf16) for A tiles: 80B, 16B-aligned, conflict-free
#define LDBT   40      // row stride (bf16) for transposed B tiles

// bit-reverse of 2-bit group index: maps col-group {0,1,2,3} -> {0,2,1,3}.
// Storing A with middle groups swapped makes each lane's WMMA fragment
// (K in {0..7,16..23} / {8..15,24..31}) contiguous in LDS.
__device__ __forceinline__ int permg(int g) { return ((g & 1) << 1) | (g >> 1); }

// load 16 contiguous bf16 (32B, 16B-aligned) as a WMMA fragment: 2x ds_load_b128
__device__ __forceinline__ v16bf ld_frag(const __bf16* p) {
    v8bf lo = *reinterpret_cast<const v8bf*>(p);
    v8bf hi = *reinterpret_cast<const v8bf*>(p + 8);
    return __builtin_shufflevector(lo, hi, 0, 1, 2, 3, 4, 5, 6, 7,
                                   8, 9, 10, 11, 12, 13, 14, 15);
}

// ---- Kernel 1: y = x @ W1 + b1  (f32 in, bf16 out) ------------------------
__global__ __launch_bounds__(256) void ssa_gemm1(const float* __restrict__ x,
                                                 const float* __restrict__ W1,
                                                 const float* __restrict__ b1,
                                                 __bf16* __restrict__ y) {
    __shared__ __bf16 As[2][TILE_M * LDA];   // 2 x 10 KB
    __shared__ __bf16 Bs[2][TILE_N * LDBT];  // 2 x 10 KB (transposed: [col][k])

    const int m0   = blockIdx.x * TILE_M;
    const int n0   = blockIdx.y * TILE_N;
    const int tid  = threadIdx.x;
    const int lane = tid & 31;
    const int wave = tid >> 5;           // 0..7
    const int wm   = (wave & 3) * 32;    // wave row base within tile
    const int wn   = (wave >> 2) * 64;   // wave col base within tile
    const int fsel = (lane >> 4) * 16;   // fragment half select (bytes*2 offset)
    const int frow = lane & 15;

    v8f acc[2][4] = {};

    // stage tile kt into buffer buf
    auto stage = [&](int buf, int k0) {
#pragma unroll
        for (int it = 0; it < 4; ++it) {           // A: 128x32 = 1024 float4
            int idx = tid + it * 256;
            int row = idx >> 3;
            int c4  = (idx & 7) * 4;
            const float4 v = *reinterpret_cast<const float4*>(
                x + (size_t)(m0 + row) * CIN + k0 + c4);
            int pc = permg(c4 >> 3) * 8 + (c4 & 7);
            __bf16* dst = &As[buf][(size_t)row * LDA + pc];
            dst[0] = (__bf16)v.x; dst[1] = (__bf16)v.y;
            dst[2] = (__bf16)v.z; dst[3] = (__bf16)v.w;
        }
#pragma unroll
        for (int it = 0; it < 4; ++it) {           // B: 32x128 = 1024 float4
            int idx = tid + it * 256;
            int row = idx >> 5;                    // k within tile
            int c4  = (idx & 31) * 4;              // col within tile
            const float4 v = *reinterpret_cast<const float4*>(
                W1 + (size_t)(k0 + row) * NFK + n0 + c4);
            __bf16* dst = &Bs[buf][(size_t)c4 * LDBT + permg(row >> 3) * 8 + (row & 7)];
            dst[0 * LDBT] = (__bf16)v.x; dst[1 * LDBT] = (__bf16)v.y;
            dst[2 * LDBT] = (__bf16)v.z; dst[3 * LDBT] = (__bf16)v.w;
        }
    };

    stage(0, 0);
    __syncthreads();

    for (int kt = 0; kt < KT; ++kt) {
        int cur = kt & 1;
        if (kt + 1 < KT) stage(cur ^ 1, (kt + 1) * TILE_K);

        v16bf af0 = ld_frag(&As[cur][(size_t)(wm + frow) * LDA + fsel]);
        v16bf af1 = ld_frag(&As[cur][(size_t)(wm + 16 + frow) * LDA + fsel]);
#pragma unroll
        for (int tn = 0; tn < 4; ++tn) {
            v16bf bf = ld_frag(&Bs[cur][(size_t)(wn + tn * 16 + frow) * LDBT + fsel]);
            acc[0][tn] = __builtin_amdgcn_wmma_f32_16x16x32_bf16(
                false, af0, false, bf, (short)0, acc[0][tn], false, false);
            acc[1][tn] = __builtin_amdgcn_wmma_f32_16x16x32_bf16(
                false, af1, false, bf, (short)0, acc[1][tn], false, false);
        }
        __syncthreads();
    }

    // epilogue: +b1, bf16 store. C layout: VGPR r -> M=r (lanes 0..15), M=r+8 (lanes 16..31)
    const int rsel = (lane >> 4) * 8;
    const int csel = lane & 15;
#pragma unroll
    for (int tm = 0; tm < 2; ++tm)
#pragma unroll
        for (int tn = 0; tn < 4; ++tn) {
            int col = n0 + wn + tn * 16 + csel;
            float bias = b1[col];
#pragma unroll
            for (int r = 0; r < 8; ++r) {
                int row = m0 + wm + tm * 16 + rsel + r;
                y[(size_t)row * NFK + col] = (__bf16)(acc[tm][tn][r] + bias);
            }
        }
}

// ---- shift source mapping (per output (h,w), channel quarter q) -----------
__device__ __forceinline__ void shift_src(int q, int h, int w,
                                          int& h1, int& w1, int& h2, int& w2) {
    h1 = h; w1 = w; h2 = h; w2 = w;
    if (q == 0)      { w1 = w > 0 ? w - 1 : 0;       h2 = h > 0 ? h - 1 : 0; }
    else if (q == 1) { w1 = w < WW-1 ? w + 1 : WW-1; h2 = h < HH-1 ? h + 1 : HH-1; }
    else if (q == 2) { h1 = h > 0 ? h - 1 : 0;       w2 = w > 0 ? w - 1 : 0; }
    else             { h1 = h < HH-1 ? h + 1 : HH-1; w2 = w < WW-1 ? w + 1 : WW-1; }
}

// ---- Kernel 2: a_part[slice][b][c] = partial sum over hw of shifted chunks -
__global__ __launch_bounds__(128) void ssa_reduce_a(const __bf16* __restrict__ y,
                                                    float* __restrict__ a_part) {
    const int b  = blockIdx.x;                    // 0..15
    const int c  = blockIdx.y * 128 + threadIdx.x;// 0..383
    const int sl = blockIdx.z;                    // 0..7
    const int q  = c / 96;
    const __bf16* yb = y + (size_t)b * HW * NFK;

    float s = 0.f;
    const int hw0 = sl * (HW / 8);
    for (int hw = hw0; hw < hw0 + HW / 8; ++hw) {
        int h = hw / WW, w = hw - h * WW;
        int h1, w1, h2, w2;
        shift_src(q, h, w, h1, w1, h2, w2);
        s += (float)yb[(size_t)(h1 * WW + w1) * NFK + c];
        s += (float)yb[(size_t)(h2 * WW + w2) * NFK + NF + c];
        s += (float)yb[(size_t)hw * NFK + 2 * NF + c];
    }
    a_part[((size_t)sl * 16 + b) * NF + c] = s;
}

// ---- Kernel 3: gating MLP + softmax over K=3 ------------------------------
__global__ __launch_bounds__(384) void ssa_gating(const float* __restrict__ a_part,
                                                  const float* __restrict__ Wg1,
                                                  const float* __restrict__ Wg2,
                                                  float* __restrict__ a_bar) {
    __shared__ float av[NF];
    __shared__ float g[NF];
    const int b = blockIdx.x;
    const int c = threadIdx.x;

    float s = 0.f;
#pragma unroll
    for (int sl = 0; sl < 8; ++sl) s += a_part[((size_t)sl * 16 + b) * NF + c];
    av[c] = s;
    __syncthreads();

    float acc = 0.f;
    for (int j = 0; j < NF; ++j) acc += av[j] * Wg1[(size_t)j * NF + c];
    // gelu (tanh approximation, jax.nn.gelu default)
    float t = 0.7978845608028654f * (acc + 0.044715f * acc * acc * acc);
    g[c] = 0.5f * acc * (1.0f + tanhf(t));
    __syncthreads();

    float hat[3];
#pragma unroll
    for (int k = 0; k < 3; ++k) {
        float hsum = 0.f;
        for (int j = 0; j < NF; ++j) hsum += g[j] * Wg2[(size_t)j * NFK + k * NF + c];
        hat[k] = hsum;
    }
    float mx  = fmaxf(hat[0], fmaxf(hat[1], hat[2]));
    float e0  = __expf(hat[0] - mx), e1 = __expf(hat[1] - mx), e2 = __expf(hat[2] - mx);
    float inv = 1.0f / (e0 + e1 + e2);
    a_bar[((size_t)b * 3 + 0) * NF + c] = e0 * inv;
    a_bar[((size_t)b * 3 + 1) * NF + c] = e1 * inv;
    a_bar[((size_t)b * 3 + 2) * NF + c] = e2 * inv;
}

// ---- Kernel 4: out = (Σ_k a_bar·xs_k) @ W2 + b2  (combine fused into A stage)
__global__ __launch_bounds__(256) void ssa_gemm2(const __bf16* __restrict__ y,
                                                 const float* __restrict__ a_bar,
                                                 const float* __restrict__ W2,
                                                 const float* __restrict__ b2,
                                                 float* __restrict__ out) {
    __shared__ __bf16 As[2][TILE_M * LDA];
    __shared__ __bf16 Bs[2][TILE_N * LDBT];

    const int m0   = blockIdx.x * TILE_M;
    const int n0   = blockIdx.y * TILE_N;
    const int tid  = threadIdx.x;
    const int lane = tid & 31;
    const int wave = tid >> 5;
    const int wm   = (wave & 3) * 32;
    const int wn   = (wave >> 2) * 64;
    const int fsel = (lane >> 4) * 16;
    const int frow = lane & 15;

    v8f acc[2][4] = {};

    auto stage = [&](int buf, int k0) {
        // A: gather 3 shifted y values per element, weight by a_bar -> bf16
#pragma unroll
        for (int it = 0; it < 16; ++it) {
            int idx = tid + it * 256;      // 0..4095
            int row = idx >> 5;            // 0..127
            int kk  = idx & 31;            // 0..31
            int tkn = m0 + row;
            int b   = tkn / HW;
            int hw  = tkn - b * HW;
            int h   = hw / WW, w = hw - h * WW;
            int c   = k0 + kk;
            int h1, w1, h2, w2;
            shift_src(c / 96, h, w, h1, w1, h2, w2);
            const __bf16* yb = y + (size_t)b * HW * NFK;
            const float*  ab = a_bar + (size_t)b * 3 * NF;
            float v = ab[c]          * (float)yb[(size_t)(h1 * WW + w1) * NFK + c]
                    + ab[NF + c]     * (float)yb[(size_t)(h2 * WW + w2) * NFK + NF + c]
                    + ab[2 * NF + c] * (float)yb[(size_t)hw * NFK + 2 * NF + c];
            int pc = permg(kk >> 3) * 8 + (kk & 7);
            As[buf][(size_t)row * LDA + pc] = (__bf16)v;
        }
        // B: W2 32x128, store transposed
#pragma unroll
        for (int it = 0; it < 4; ++it) {
            int idx = tid + it * 256;
            int row = idx >> 5;
            int c4  = (idx & 31) * 4;
            const float4 v = *reinterpret_cast<const float4*>(
                W2 + (size_t)(k0 + row) * NF + n0 + c4);
            __bf16* dst = &Bs[buf][(size_t)c4 * LDBT + permg(row >> 3) * 8 + (row & 7)];
            dst[0 * LDBT] = (__bf16)v.x; dst[1 * LDBT] = (__bf16)v.y;
            dst[2 * LDBT] = (__bf16)v.z; dst[3 * LDBT] = (__bf16)v.w;
        }
    };

    stage(0, 0);
    __syncthreads();

    for (int kt = 0; kt < KT; ++kt) {
        int cur = kt & 1;
        if (kt + 1 < KT) stage(cur ^ 1, (kt + 1) * TILE_K);

        v16bf af0 = ld_frag(&As[cur][(size_t)(wm + frow) * LDA + fsel]);
        v16bf af1 = ld_frag(&As[cur][(size_t)(wm + 16 + frow) * LDA + fsel]);
#pragma unroll
        for (int tn = 0; tn < 4; ++tn) {
            v16bf bf = ld_frag(&Bs[cur][(size_t)(wn + tn * 16 + frow) * LDBT + fsel]);
            acc[0][tn] = __builtin_amdgcn_wmma_f32_16x16x32_bf16(
                false, af0, false, bf, (short)0, acc[0][tn], false, false);
            acc[1][tn] = __builtin_amdgcn_wmma_f32_16x16x32_bf16(
                false, af1, false, bf, (short)0, acc[1][tn], false, false);
        }
        __syncthreads();
    }

    const int rsel = (lane >> 4) * 8;
    const int csel = lane & 15;
#pragma unroll
    for (int tm = 0; tm < 2; ++tm)
#pragma unroll
        for (int tn = 0; tn < 4; ++tn) {
            int col = n0 + wn + tn * 16 + csel;
            float bias = b2[col];
#pragma unroll
            for (int r = 0; r < 8; ++r) {
                int row = m0 + wm + tm * 16 + rsel + r;
                // final output is never re-read on device: non-temporal store
                __builtin_nontemporal_store(acc[tm][tn][r] + bias,
                                            &out[(size_t)row * NF + col]);
            }
        }
}

// ---------------------------------------------------------------------------
extern "C" void kernel_launch(void* const* d_in, const int* in_sizes, int n_in,
                              void* d_out, int out_size, void* d_ws, size_t ws_size,
                              hipStream_t stream) {
    const float* x   = (const float*)d_in[0];
    const float* W1  = (const float*)d_in[1];
    const float* b1  = (const float*)d_in[2];
    const float* Wg1 = (const float*)d_in[3];
    const float* Wg2 = (const float*)d_in[4];
    const float* W2  = (const float*)d_in[5];
    const float* b2  = (const float*)d_in[6];
    float* out = (float*)d_out;

    char* ws = (char*)d_ws;
    const size_t y_bytes = (size_t)NTOK * NFK * sizeof(__bf16);   // ~110 MB
    __bf16* y      = (__bf16*)ws;
    float*  a_part = (float*)(ws + y_bytes);                      // 8*16*384 f32
    float*  a_bar  = (float*)(ws + y_bytes + 8 * 16 * NF * sizeof(float));

    dim3 g1(NTOK / TILE_M, NFK / TILE_N);   // (392, 9)
    ssa_gemm1<<<g1, 256, 0, stream>>>(x, W1, b1, y);

    dim3 g2(16, NF / 128, 8);               // (16, 3, 8)
    ssa_reduce_a<<<g2, 128, 0, stream>>>(y, a_part);

    ssa_gating<<<16, NF, 0, stream>>>(a_part, Wg1, Wg2, a_bar);

    dim3 g4(NTOK / TILE_M, NF / TILE_N);    // (392, 3)
    ssa_gemm2<<<g4, 256, 0, stream>>>(y, a_bar, W2, b2, out);
}
